// RegimePatternBank_54992761258654
// MI455X (gfx1250) — matrix operand order repaired and checked by reading, not compiled, not run on hardware
//
#include <hip/hip_runtime.h>
#include <hip/hip_bf16.h>

typedef _Float16 v16h __attribute__((ext_vector_type(16)));
typedef _Float16 v8h  __attribute__((ext_vector_type(8)));
typedef _Float16 v4h  __attribute__((ext_vector_type(4)));
typedef float    v8f  __attribute__((ext_vector_type(8)));
typedef float    v4f  __attribute__((ext_vector_type(4)));
typedef unsigned int u32x4 __attribute__((ext_vector_type(4)));
typedef int          i32x4 __attribute__((ext_vector_type(4)));
typedef int          i32x8 __attribute__((ext_vector_type(8)));

#define B_DIM 32768
#define D_DIM 512
#define P_DIM 4096
#define WAVES 4
#define NT_PER 2                      // 16-col tiles per TDM transfer
#define BROWS (NT_PER * 16)           // prototype rows per TDM tile
#define BPITCH 520                    // halves per row in LDS (1040 B)
#define ITERS (P_DIM / (16 * NT_PER)) // 128

__device__ __forceinline__ v16h join8(v8h lo, v8h hi) {
  return __builtin_shufflevector(lo, hi, 0,1,2,3,4,5,6,7,8,9,10,11,12,13,14,15);
}
__device__ __forceinline__ v16h ldB(const _Float16* p) {
  return join8(*(const v8h*)p, *(const v8h*)(p + 16));
}
__device__ __forceinline__ unsigned umx(unsigned a, unsigned b){ return a > b ? a : b; }
__device__ __forceinline__ unsigned umn(unsigned a, unsigned b){ return a < b ? a : b; }

// TDM: DMA a [BROWS x 1024B] row-major tile from global into LDS with a 16B
// pad after each 1024B row (LDS pitch 1040B).  D# per CDNA5 ISA ch.8.
__device__ __forceinline__ void tdm_load_tile(const _Float16* gsrc, unsigned lds_off) {
  unsigned long long ga = (unsigned long long)(const void*)gsrc;
  u32x4 g0;
  g0[0] = 1u;                                                // count=1, no gather
  g0[1] = lds_off;                                           // lds_addr (bytes)
  g0[2] = (unsigned)ga;                                      // global_addr[31:0]
  g0[3] = (unsigned)((ga >> 32) & 0x01FFFFFFu) | (2u << 30); // addr[56:32] | type=2
  i32x8 g1;
  g1[0] = (3 << 16)            // data_size = 8B
        | (1 << 20)            // pad_enable
        | (7 << 22)            // pad_interval: 256 DWORDs (1024B)
        | (3 << 25);           // pad_amount: 4 DWORDs (16B)
  g1[1] = (128 << 16);         // tensor_dim0[15:0] = 128 (8B units)
  g1[2] = (BROWS << 16);       // tensor_dim1[15:0] = BROWS
  g1[3] = (128 << 16);         // tile_dim0 = 128
  g1[4] = BROWS;               // tile_dim1 = BROWS, tile_dim2 = 0
  g1[5] = 128;                 // tensor_dim0_stride[31:0] = 128 (8B units)
  g1[6] = 0;
  g1[7] = 0;
  i32x4 z4 = {0, 0, 0, 0};
#if __clang_major__ >= 23
  i32x8 z8 = {0, 0, 0, 0, 0, 0, 0, 0};
  __builtin_amdgcn_tensor_load_to_lds(g0, g1, z4, z4, z8, 0);
#else
  __builtin_amdgcn_tensor_load_to_lds(g0, g1, z4, z4, 0);
#endif
}

// ---------------------------------------------------------------------------
// Prep: per-row L2 normalize + f32->f16 (normalized and raw copies).
// ---------------------------------------------------------------------------
__global__ void row_norm_f16(const float* __restrict__ src,
                             _Float16* __restrict__ normDst,
                             _Float16* __restrict__ rawDst,
                             int rows) {
  int wave = blockIdx.x * (blockDim.x >> 5) + (threadIdx.x >> 5);
  if (wave >= rows) return;
  int lane = threadIdx.x & 31;
  const v4f* r = (const v4f*)(src + (size_t)wave * D_DIM);
  v4f x[4];
  float ss = 0.0f;
#pragma unroll
  for (int i = 0; i < 4; ++i) {
    x[i] = r[lane + 32 * i];
    ss += x[i][0]*x[i][0] + x[i][1]*x[i][1] + x[i][2]*x[i][2] + x[i][3]*x[i][3];
  }
#pragma unroll
  for (int off = 16; off >= 1; off >>= 1) ss += __shfl_xor(ss, off, 32);
  float scale = 1.0f / fmaxf(sqrtf(ss), 1e-12f);
  v4h* nd = (v4h*)(normDst + (size_t)wave * D_DIM);
  v4h* rd = (v4h*)(rawDst  + (size_t)wave * D_DIM);
#pragma unroll
  for (int i = 0; i < 4; ++i) {
    v4h hn, hr;
#pragma unroll
    for (int j = 0; j < 4; ++j) {
      hn[j] = (_Float16)(x[i][j] * scale);
      hr[j] = (_Float16)(x[i][j]);
    }
    nd[lane + 32 * i] = hn;
    rd[lane + 32 * i] = hr;
  }
}

__global__ void conv_weights(const float* __restrict__ W1, const float* __restrict__ W2,
                             _Float16* __restrict__ W1H, _Float16* __restrict__ W2H) {
  int i = threadIdx.x + blockIdx.x * blockDim.x;
  int stride = blockDim.x * gridDim.x;
  for (int t = i; t < 32 * 1024; t += stride) W1H[t] = (_Float16)W1[t];
  for (int t = i; t < 16 * 32;   t += stride) W2H[t] = (_Float16)W2[t];
}

// ---------------------------------------------------------------------------
// Main fused kernel.
// ---------------------------------------------------------------------------
struct alignas(64) WaveScratch {
  union {
    unsigned mergeK[16 * 16 * 3];
    _Float16 hbuf[48 * 32];
  } u;
  unsigned wIdx[16 * 3];
  float    wVal[16 * 3];
};

__launch_bounds__(WAVES * 32)
__global__ void rpb_main(const _Float16* __restrict__ regN,
                         const _Float16* __restrict__ regR,
                         const _Float16* __restrict__ proN,
                         const _Float16* __restrict__ proR,
                         const _Float16* __restrict__ W1H,
                         const _Float16* __restrict__ W2H,
                         const float* __restrict__ b1,
                         const float* __restrict__ b2,
                         float* __restrict__ out) {
  __shared__ __align__(16) _Float16 Bbuf[2][BROWS * BPITCH];
  __shared__ WaveScratch lds[WAVES];

  const int wv   = threadIdx.x >> 5;
  const int lane = threadIdx.x & 31;
  const int mr   = lane & 15;
  const int kh   = lane >> 4;
  WaveScratch& W = lds[wv];
  const int row0 = (blockIdx.x * WAVES + wv) * 16;

  // ---- preload all 16 A fragments (K = 512) into registers ----
  v16h aF[16];
  {
    const _Float16* abase = regN + (size_t)(row0 + mr) * D_DIM + kh * 8;
#pragma unroll
    for (int kt = 0; kt < 16; ++kt)
      aF[kt] = join8(*(const v8h*)(abase + kt * 32),
                     *(const v8h*)(abase + kt * 32 + 16));
  }

  unsigned t0[8], t1[8], t2[8];
#pragma unroll
  for (int v = 0; v < 8; ++v) { t0[v] = 0u; t1[v] = 0u; t2[v] = 0u; }

  const unsigned buf0 = (unsigned)(uintptr_t)&Bbuf[0][0];
  const unsigned buf1 = (unsigned)(uintptr_t)&Bbuf[1][0];

  if (wv == 0) tdm_load_tile(proN, buf0);

  for (int it = 0; it < ITERS; ++it) {
    if (wv == 0) __builtin_amdgcn_s_wait_tensorcnt(0);   // tile `it` arrived
    __syncthreads();
    if (wv == 0 && it + 1 < ITERS)
      tdm_load_tile(proN + (size_t)(it + 1) * BROWS * D_DIM,
                    ((it + 1) & 1) ? buf1 : buf0);

    // Flattened, software-pipelined pass over NT_PER*16 B fragments:
    // ds_loads issued 2 fragments (~2 WMMAs) ahead of use.
    const _Float16* bb = &Bbuf[it & 1][0] + (size_t)mr * BPITCH + kh * 8;
    const _Float16* f0 = bb;                     // frag f: tile j=f>>4, kt=f&15
    v16h bA = ldB(f0);
    v16h bB = ldB(f0 + 32);
    v8f acc = {};
#pragma unroll
    for (int f = 0; f < NT_PER * 16; ++f) {
      v16h nxt = bB;
      if (f + 2 < NT_PER * 16) {
        int g = f + 2;
        nxt = ldB(bb + (size_t)(g >> 4) * (16 * BPITCH) + (g & 15) * 32);
      }
      acc = __builtin_amdgcn_wmma_f32_16x16x32_f16(false, aF[f & 15], false, bA,
                                                   (short)0, acc, false, false);
      bA = bB; bB = nxt;
      if ((f & 15) == 15) {
        unsigned p = (unsigned)((it * NT_PER + (f >> 4)) * 16 + mr);
#pragma unroll
        for (int v = 0; v < 8; ++v) {
          unsigned k = (__float_as_uint(acc[v] + 2.0f) & 0xFFFFF000u) | p;
          unsigned a0 = t0[v]; t0[v] = umx(a0, k); k = umn(a0, k);
          unsigned a1 = t1[v]; t1[v] = umx(a1, k); k = umn(a1, k);
          t2[v] = umx(t2[v], k);
        }
        acc = (v8f){};
      }
    }
  }

  // ---- cross-lane merge via LDS ----
#pragma unroll
  for (int v = 0; v < 8; ++v) {
    int m = v + kh * 8;
    W.u.mergeK[(m * 16 + mr) * 3 + 0] = t0[v];
    W.u.mergeK[(m * 16 + mr) * 3 + 1] = t1[v];
    W.u.mergeK[(m * 16 + mr) * 3 + 2] = t2[v];
  }
  __syncthreads();

  if (lane < 16) {
    unsigned g0 = 0u, g1 = 0u, g2 = 0u;
    for (int s = 0; s < 16; ++s) {
#pragma unroll
      for (int j = 0; j < 3; ++j) {
        unsigned k = W.u.mergeK[(lane * 16 + s) * 3 + j];
        unsigned a0 = g0; g0 = umx(a0, k); k = umn(a0, k);
        unsigned a1 = g1; g1 = umx(a1, k); k = umn(a1, k);
        g2 = umx(g2, k);
      }
    }
    float s0 = __uint_as_float(g0 & 0xFFFFF000u) - 2.0f;
    float s1 = __uint_as_float(g1 & 0xFFFFF000u) - 2.0f;
    float s2 = __uint_as_float(g2 & 0xFFFFF000u) - 2.0f;
    float e1 = __expf(5.0f * (s1 - s0));
    float e2 = __expf(5.0f * (s2 - s0));
    float inv = 1.0f / (1.0f + e1 + e2);
    W.wIdx[lane * 3 + 0] = g0 & 0xFFFu;
    W.wIdx[lane * 3 + 1] = g1 & 0xFFFu;
    W.wIdx[lane * 3 + 2] = g2 & 0xFFFu;
    W.wVal[lane * 3 + 0] = inv;
    W.wVal[lane * 3 + 1] = e1 * inv;
    W.wVal[lane * 3 + 2] = e2 * inv;
  }
  __syncthreads();

  // ---- MLP layer 1 via WMMA: [16,1024] x [1024,32] per candidate ----
  const float b1a = b1[mr];
  const float b1b = b1[16 + mr];
  const _Float16* rg = regR + (size_t)(row0 + mr) * D_DIM + kh * 8;
#pragma unroll 1
  for (int kc = 0; kc < 3; ++kc) {
    unsigned pidx = W.wIdx[mr * 3 + kc];
    const _Float16* pr = proR + (size_t)pidx * D_DIM + kh * 8;
    v8f acc0 = {}, acc1 = {};
#pragma unroll 4
    for (int kt = 0; kt < 32; ++kt) {
      const _Float16* ap = (kt < 16) ? (rg + kt * 32) : (pr + (kt - 16) * 32);
      v16h aF2 = join8(*(const v8h*)ap, *(const v8h*)(ap + 16));
      const _Float16* w1p = W1H + (size_t)mr * 1024 + kh * 8 + kt * 32;
      v16h bF0 = join8(*(const v8h*)w1p, *(const v8h*)(w1p + 16));
      const _Float16* w1q = w1p + 16 * 1024;
      v16h bF1 = join8(*(const v8h*)w1q, *(const v8h*)(w1q + 16));
      acc0 = __builtin_amdgcn_wmma_f32_16x16x32_f16(false, aF2, false, bF0,
                                                    (short)0, acc0, false, false);
      acc1 = __builtin_amdgcn_wmma_f32_16x16x32_f16(false, aF2, false, bF1,
                                                    (short)0, acc1, false, false);
    }
#pragma unroll
    for (int v = 0; v < 8; ++v) {
      int hr = kc * 16 + v + kh * 8;
      W.u.hbuf[hr * 32 + mr]      = (_Float16)fmaxf(acc0[v] + b1a, 0.0f);
      W.u.hbuf[hr * 32 + 16 + mr] = (_Float16)fmaxf(acc1[v] + b1b, 0.0f);
    }
  }
  __syncthreads();

  // ---- MLP layer 2 via WMMA + softmax-weighted sum ----
  const _Float16* w2p = W2H + mr * 32 + kh * 8;
  v16h bW2 = join8(*(const v8h*)w2p, *(const v8h*)(w2p + 16));
  const float b2o = b2[mr];
  v8f oacc = {};
#pragma unroll
  for (int kc = 0; kc < 3; ++kc) {
    const _Float16* hp = &W.u.hbuf[(kc * 16 + mr) * 32 + kh * 8];
    v16h aH = join8(*(const v8h*)hp, *(const v8h*)(hp + 16));
    v8f f = {};
    f = __builtin_amdgcn_wmma_f32_16x16x32_f16(false, aH, false, bW2,
                                               (short)0, f, false, false);
#pragma unroll
    for (int v = 0; v < 8; ++v) {
      int m = v + kh * 8;
      oacc[v] += W.wVal[m * 3 + kc] * (f[v] + b2o);
    }
  }
#pragma unroll
  for (int v = 0; v < 8; ++v) {
    int m = v + kh * 8;
    out[(size_t)(row0 + m) * 16 + mr] = oacc[v];
  }
}

// ---------------------------------------------------------------------------
extern "C" void kernel_launch(void* const* d_in, const int* in_sizes, int n_in,
                              void* d_out, int out_size, void* d_ws, size_t ws_size,
                              hipStream_t stream) {
  const float* regime = (const float*)d_in[0];
  const float* protos = (const float*)d_in[1];
  const float* W1     = (const float*)d_in[2];
  const float* b1     = (const float*)d_in[3];
  const float* W2     = (const float*)d_in[4];
  const float* b2     = (const float*)d_in[5];
  float* out = (float*)d_out;

  char* ws = (char*)d_ws;
  _Float16* regN = (_Float16*)ws; ws += (size_t)B_DIM * D_DIM * 2;
  _Float16* regR = (_Float16*)ws; ws += (size_t)B_DIM * D_DIM * 2;
  _Float16* proN = (_Float16*)ws; ws += (size_t)P_DIM * D_DIM * 2;
  _Float16* proR = (_Float16*)ws; ws += (size_t)P_DIM * D_DIM * 2;
  _Float16* W1H  = (_Float16*)ws; ws += 32 * 1024 * 2;
  _Float16* W2H  = (_Float16*)ws; ws += 16 * 32 * 2;

  row_norm_f16<<<B_DIM / 8, 256, 0, stream>>>(regime, regN, regR, B_DIM);
  row_norm_f16<<<P_DIM / 8, 256, 0, stream>>>(protos, proN, proR, P_DIM);
  conv_weights<<<64, 256, 0, stream>>>(W1, W2, W1H, W2H);
  rpb_main<<<B_DIM / (16 * WAVES), WAVES * 32, 0, stream>>>(
      regN, regR, proN, proR, W1H, W2H, b1, b2, out);
}
